// EdgeAwareBiLSTM_42614665511527
// MI455X (gfx1250) — matrix operand order repaired and checked by reading, not compiled, not run on hardware
//
#include <hip/hip_runtime.h>

#define B_   32
#define L_   16384
#define HID_ 32
#define G_   128   // 4*HID

typedef __attribute__((ext_vector_type(16))) __bf16 v16bf;
typedef __attribute__((ext_vector_type(8)))  __bf16 v8bf;
typedef __attribute__((ext_vector_type(8)))  float  v8f;

__device__ __forceinline__ v8f wmma_bf16(v16bf a, v16bf b, v8f c) {
  // D = A(16x32 bf16) x B(32x16 bf16) + C(16x16 f32)
  return __builtin_amdgcn_wmma_f32_16x16x32_bf16(false, a, false, b, (short)0, c, false, false);
}

__device__ __forceinline__ v16bf mk16(v8bf lo, v8bf hi) {
  v16bf r;
#pragma unroll
  for (int i = 0; i < 8; i++) { r[i] = lo[i]; r[i + 8] = hi[i]; }
  return r;
}

__device__ __forceinline__ float sigm(float x) {
  return __builtin_amdgcn_rcpf(1.0f + __expf(-x));
}
__device__ __forceinline__ float tanh_(float x) {
  return 2.0f * sigm(2.0f * x) - 1.0f;
}

// ---------------------------------------------------------------------------
// Kernel 1: projection  x(B*L,64) @ proj_W.T(64,32) + b  -> bf16 (B*L,32)
// x[b,t,c] = features[b,c,t]
// ---------------------------------------------------------------------------
__global__ void proj_kernel(const float* __restrict__ feat,
                            const float* __restrict__ pW,
                            const float* __restrict__ pb,
                            __bf16* __restrict__ proj) {
  const int lane  = threadIdx.x & 31;
  const int n     = lane & 15;
  const int kb    = (lane < 16) ? 0 : 8;    // A-layout K base
  const int koffB = (lane < 16) ? 0 : 16;   // unused; B tiles also use kb

  v16bf Bt[2][2];
  float cb[2];
#pragma unroll
  for (int nt = 0; nt < 2; nt++) {
    const float* wr = pW + (size_t)(nt * 16 + n) * 64;
    cb[nt] = pb[nt * 16 + n];
#pragma unroll
    for (int kc = 0; kc < 2; kc++) {
      v16bf bb;
#pragma unroll
      for (int j = 0; j < 16; j++) bb[j] = (__bf16)wr[kc * 32 + kb + (j & 7) + ((j >> 3) << 4)];
      Bt[nt][kc] = bb;
    }
  }

  const int NT = (B_ * L_) / 16;
  int wave = blockIdx.x * (blockDim.x >> 5) + (threadIdx.x >> 5);
  int nw   = gridDim.x * (blockDim.x >> 5);
  for (int rt = wave; rt < NT; rt += nw) {
    int r0 = rt * 16;
    int b  = r0 / L_;
    int t0 = r0 - b * L_;
    int t  = t0 + n;  // row this lane supplies for the A matrix

    v16bf A[2];
#pragma unroll
    for (int kc = 0; kc < 2; kc++) {
      v16bf a;
#pragma unroll
      for (int i = 0; i < 8; i++) {
        int c0 = kc * 32 + kb + i;
        int c1 = kc * 32 + kb + 16 + i;
        a[i]     = (__bf16)feat[((size_t)(b * 64 + c0)) * L_ + t];
        a[i + 8] = (__bf16)feat[((size_t)(b * 64 + c1)) * L_ + t];
      }
      A[kc] = a;
    }
    const int mbase = (lane < 16) ? 0 : 8;
#pragma unroll
    for (int nt = 0; nt < 2; nt++) {
      v8f c;
#pragma unroll
      for (int i = 0; i < 8; i++) c[i] = cb[nt];
      v8f d = wmma_bf16(A[1], Bt[nt][1], wmma_bf16(A[0], Bt[nt][0], c));
#pragma unroll
      for (int r = 0; r < 8; r++)
        proj[(size_t)(r0 + mbase + r) * 32 + nt * 16 + n] = (__bf16)d[r];
    }
  }
}

// ---------------------------------------------------------------------------
// Kernel 2: BiLSTM. 4 single-wave blocks: (dir, batch-half). No inter-wave
// sync in the 16384-step serial loop; weights live in VGPRs as bf16 B-tiles.
// x_t A-tile is software-pipelined one step ahead so its global load and the
// 8 x-side WMMAs overlap the h-side LDS transpose on the serial chain.
// ---------------------------------------------------------------------------
__global__ void __launch_bounds__(32, 1)
lstm_kernel(const __bf16* __restrict__ proj,
            const float* __restrict__ Wih_f, const float* __restrict__ Whh_f,
            const float* __restrict__ bih_f, const float* __restrict__ bhh_f,
            const float* __restrict__ Wih_b, const float* __restrict__ Whh_b,
            const float* __restrict__ bih_b, const float* __restrict__ bhh_b,
            __bf16* __restrict__ hout) {
  __shared__ __bf16 hbuf[16 * 32];  // h tile, row-major bf16, for D->A transpose
  const int lane = threadIdx.x & 31;
  const int dir  = blockIdx.x >> 1;
  const int b0   = (blockIdx.x & 1) * 16;
  const float* Wih = dir ? Wih_b : Wih_f;
  const float* Whh = dir ? Whh_b : Whh_f;
  const float* bih = dir ? bih_b : bih_f;
  const float* bhh = dir ? bhh_b : bhh_f;

  const int n     = lane & 15;
  const int kb    = (lane < 16) ? 0 : 8;
  const int koffB = (lane < 16) ? 0 : 16;
  const int mbase = (lane < 16) ? 0 : 8;

  // Register-resident bf16 weight tiles (8 N-tiles each) + fused bias
  v16bf Bih[8], Bhh[8];
  float bias[8];
#pragma unroll
  for (int nt = 0; nt < 8; nt++) {
    int g = nt * 16 + n;
    bias[nt] = bih[g] + bhh[g];
    const float* r1 = Wih + (size_t)g * 32;
    const float* r2 = Whh + (size_t)g * 32;
    v16bf t1, t2;
#pragma unroll
    for (int j = 0; j < 16; j++) {
      t1[j] = (__bf16)r1[kb + (j & 7) + ((j >> 3) << 4)];
      t2[j] = (__bf16)r2[kb + (j & 7) + ((j >> 3) << 4)];
    }
    Bih[nt] = t1;
    Bhh[nt] = t2;
  }

#pragma unroll
  for (int i = 0; i < 16; i++) hbuf[lane + 32 * i] = (__bf16)0.0f;  // h0 = 0
  float cst[16];
#pragma unroll
  for (int i = 0; i < 16; i++) cst[i] = 0.0f;                       // c0 = 0

  const __bf16* xbase = proj + (size_t)(b0 + n) * L_ * 32;
  // prologue: prefetch x A-tile for the first step
  int t_first = dir ? (L_ - 1) : 0;
  v16bf Ax = mk16(*(const v8bf*)(xbase + (size_t)t_first * 32 + kb),
                  *(const v8bf*)(xbase + (size_t)t_first * 32 + kb + 16));

  for (int s = 0; s < L_; s++) {
    int t  = dir ? (L_ - 1 - s) : s;
    // prefetch next step's x A-tile (clamped; issues a full iteration early)
    int sn = (s + 1 < L_) ? (s + 1) : s;
    int tn = dir ? (L_ - 1 - sn) : sn;
    v16bf Axn = mk16(*(const v8bf*)(xbase + (size_t)tn * 32 + kb),
                     *(const v8bf*)(xbase + (size_t)tn * 32 + kb + 16));

    // h_{t-1} A-tile from LDS (row-major bf16)
    const __bf16* hr = hbuf + n * 32;
    v16bf Ah = mk16(*(const v8bf*)(hr + kb), *(const v8bf*)(hr + kb + 16));

    v8f d[8];
#pragma unroll
    for (int nt = 0; nt < 8; nt++) {
      v8f z = {0.f, 0.f, 0.f, 0.f, 0.f, 0.f, 0.f, 0.f};
      d[nt] = wmma_bf16(Ah, Bhh[nt], wmma_bf16(Ax, Bih[nt], z));  // K=64 chain
    }
    Ax = Axn;

    // gate nonlinearities + state update; tiles (0,1)=i (2,3)=f (4,5)=g (6,7)=o
#pragma unroll
    for (int cc = 0; cc < 2; cc++) {
#pragma unroll
      for (int r = 0; r < 8; r++) {
        float iv = sigm (d[0 + cc][r] + bias[0 + cc]);
        float fv = sigm (d[2 + cc][r] + bias[2 + cc]);
        float gv = tanh_(d[4 + cc][r] + bias[4 + cc]);
        float ov = sigm (d[6 + cc][r] + bias[6 + cc]);
        float cn = fv * cst[cc * 8 + r] + iv * gv;
        cst[cc * 8 + r] = cn;
        float hv = ov * tanh_(cn);
        hbuf[(mbase + r) * 32 + cc * 16 + n] = (__bf16)hv;
      }
    }

    // spill h tile to global lstm_out (coalesced b128 via the LDS copy)
#pragma unroll
    for (int half = 0; half < 2; half++) {
      int q   = lane + half * 32;
      int row = q >> 2;
      int off = (q & 3) * 8;
      v8bf v = *(const v8bf*)(hbuf + row * 32 + off);
      *(v8bf*)(hout + ((size_t)(b0 + row) * L_ + t) * 64 + dir * 32 + off) = v;
    }
  }
}

// ---------------------------------------------------------------------------
// Kernel 3: boundary classifier: relu(lstm@W1.T + b1) @ W2.T + b2 -> sigmoid
// ---------------------------------------------------------------------------
__global__ void cls_kernel(const __bf16* __restrict__ hout,
                           const float* __restrict__ W1, const float* __restrict__ b1,
                           const float* __restrict__ W2, const float* __restrict__ b2,
                           float* __restrict__ praw) {
  const int lane  = threadIdx.x & 31;
  const int n     = lane & 15;
  const int kb    = (lane < 16) ? 0 : 8;
  const int koffB = (lane < 16) ? 0 : 16;

  v16bf Bt[2][2];
  float cb[2], w2v[2];
#pragma unroll
  for (int nt = 0; nt < 2; nt++) {
    const float* wr = W1 + (size_t)(nt * 16 + n) * 64;
    cb[nt]  = b1[nt * 16 + n];
    w2v[nt] = W2[nt * 16 + n];
#pragma unroll
    for (int kc = 0; kc < 2; kc++) {
      v16bf bb;
#pragma unroll
      for (int j = 0; j < 16; j++) bb[j] = (__bf16)wr[kc * 32 + kb + (j & 7) + ((j >> 3) << 4)];
      Bt[nt][kc] = bb;
    }
  }
  const float bias2 = b2[0];

  const int NT = (B_ * L_) / 16;
  int wave = blockIdx.x * (blockDim.x >> 5) + (threadIdx.x >> 5);
  int nw   = gridDim.x * (blockDim.x >> 5);
  const int mbase = (lane < 16) ? 0 : 8;
  for (int rt = wave; rt < NT; rt += nw) {
    int r0 = rt * 16;
    const __bf16* ar = hout + (size_t)(r0 + n) * 64;
    v16bf A0 = mk16(*(const v8bf*)(ar + kb),      *(const v8bf*)(ar + kb + 16));
    v16bf A1 = mk16(*(const v8bf*)(ar + 32 + kb), *(const v8bf*)(ar + 32 + kb + 16));

    v8f d[2];
#pragma unroll
    for (int nt = 0; nt < 2; nt++) {
      v8f c;
#pragma unroll
      for (int i = 0; i < 8; i++) c[i] = cb[nt];
      d[nt] = wmma_bf16(A1, Bt[nt][1], wmma_bf16(A0, Bt[nt][0], c));
    }
#pragma unroll
    for (int r = 0; r < 8; r++) {
      float s0 = fmaxf(d[0][r], 0.0f) * w2v[0] + fmaxf(d[1][r], 0.0f) * w2v[1];
#pragma unroll
      for (int off = 1; off < 16; off <<= 1)
        s0 += __shfl_xor(s0, off, 16);     // reduce 16 cols within lane half
      if (n == r)
        praw[r0 + mbase + r] = sigm(s0 + bias2);
    }
  }
}

// ---------------------------------------------------------------------------
// Kernel 4: boundary scaling, clip, edge-adjust
// ---------------------------------------------------------------------------
__device__ __forceinline__ float pfin(const float* praw, int b, int t) {
  float v = praw[(size_t)b * L_ + t];
  if (t == 0 || t == L_ - 1) v *= 1.2f;
  return fminf(fmaxf(v, 0.0f), 1.0f);
}
__global__ void edge_kernel(const float* __restrict__ praw, float* __restrict__ out) {
  int idx = blockIdx.x * blockDim.x + threadIdx.x;
  if (idx >= B_ * L_) return;
  int b = idx / L_, t = idx - b * L_;
  float p = pfin(praw, b, t);
  float adj = 0.0f;
  if (t > 0 && t < L_ - 1) {
    float pm = pfin(praw, b, t - 1);
    float pp = pfin(praw, b, t + 1);
    float left = p - pm, right = p - pp;
    float al = fabsf(left), ar = fabsf(right);
    if (left < 0.0f && al > ar)       adj = -1.0f;
    else if (right < 0.0f && ar > al) adj = 1.0f;
  }
  out[idx] = p;
  out[(size_t)B_ * L_ + idx] = adj;
}

// ---------------------------------------------------------------------------
// Kernel 5: real_prob = sigmoid(mean_t(lstm_out) @ rc_W.T + rc_b)
// ---------------------------------------------------------------------------
__global__ void rprob_kernel(const __bf16* __restrict__ hout,
                             const float* __restrict__ rcW, const float* __restrict__ rcb,
                             float* __restrict__ out) {
  __shared__ float sums[4][64];
  int b = blockIdx.x;
  int tid = threadIdx.x;           // 256 threads
  int col = tid & 63, grp = tid >> 6;
  float acc = 0.0f;
  for (int t = grp; t < L_; t += 4)
    acc += (float)hout[((size_t)b * L_ + t) * 64 + col];
  sums[grp][col] = acc;
  __syncthreads();
  if (tid < 64) {
    float tot = sums[0][tid] + sums[1][tid] + sums[2][tid] + sums[3][tid];
    sums[0][tid] = (tot * (1.0f / (float)L_)) * rcW[tid];
  }
  __syncthreads();
  if (tid == 0) {
    float s = 0.0f;
#pragma unroll
    for (int i = 0; i < 64; i++) s += sums[0][i];
    out[(size_t)2 * B_ * L_ + b] = sigm(s + rcb[0]);
  }
}

// ---------------------------------------------------------------------------
extern "C" void kernel_launch(void* const* d_in, const int* in_sizes, int n_in,
                              void* d_out, int out_size, void* d_ws, size_t ws_size,
                              hipStream_t stream) {
  (void)in_sizes; (void)n_in; (void)out_size; (void)ws_size;
  const float* feat  = (const float*)d_in[0];
  const float* pW    = (const float*)d_in[1];
  const float* pb    = (const float*)d_in[2];
  const float* Wih_f = (const float*)d_in[3];
  const float* Whh_f = (const float*)d_in[4];
  const float* bih_f = (const float*)d_in[5];
  const float* bhh_f = (const float*)d_in[6];
  const float* Wih_b = (const float*)d_in[7];
  const float* Whh_b = (const float*)d_in[8];
  const float* bih_b = (const float*)d_in[9];
  const float* bhh_b = (const float*)d_in[10];
  const float* W1    = (const float*)d_in[11];
  const float* b1    = (const float*)d_in[12];
  const float* W2    = (const float*)d_in[13];
  const float* b2    = (const float*)d_in[14];
  const float* rcW   = (const float*)d_in[15];
  const float* rcb   = (const float*)d_in[16];

  char* ws = (char*)d_ws;
  __bf16* proj = (__bf16*)ws;                                               // 32 MB
  __bf16* hout = (__bf16*)(ws + (size_t)B_ * L_ * 32 * 2);                  // 64 MB
  float*  praw = (float*)(ws + (size_t)B_ * L_ * 32 * 2 + (size_t)B_ * L_ * 64 * 2);  // 2 MB

  proj_kernel <<<512, 256, 0, stream>>>(feat, pW, pb, proj);
  lstm_kernel <<<4, 32, 0, stream>>>(proj, Wih_f, Whh_f, bih_f, bhh_f,
                                     Wih_b, Whh_b, bih_b, bhh_b, hout);
  cls_kernel  <<<1024, 256, 0, stream>>>(hout, W1, b1, W2, b2, praw);
  edge_kernel <<<(B_ * L_ + 255) / 256, 256, 0, stream>>>(praw, (float*)d_out);
  rprob_kernel<<<B_, 256, 0, stream>>>(hout, rcW, rcb, (float*)d_out);
}
